// EGNN_45655502356935
// MI455X (gfx1250) — matrix operand order repaired and checked by reading, not compiled, not run on hardware
//
#include <hip/hip_runtime.h>
#include <hip/hip_bf16.h>

typedef __attribute__((ext_vector_type(16))) __bf16 v16bf;
typedef __attribute__((ext_vector_type(8)))  __bf16 v8bf;
typedef __attribute__((ext_vector_type(8)))  float  v8f;
typedef __attribute__((ext_vector_type(4)))  float  v4f;

#define HH   512
#define WW   512
#define CIN  128
#define OC   128
#define NCLS 16
#define NSP  16384
#define NE   262144

// ---------------- utility kernels ----------------

__global__ void k_zero(float* __restrict__ p, int n) {
  int i = blockIdx.x * blockDim.x + threadIdx.x;
  int stride = gridDim.x * blockDim.x;
  for (; i < n; i += stride) p[i] = 0.0f;
}

// one thread per (pixel, 16-channel chunk): NE*8 threads total
__global__ void k_pool_accum(const float* __restrict__ x, const int* __restrict__ seg,
                             float* __restrict__ counts, float* __restrict__ sp) {
  int t = blockIdx.x * blockDim.x + threadIdx.x;
  int p  = t >> 3;
  int c0 = (t & 7) << 4;
  int s = seg[p];
  if ((t & 7) == 0) atomicAdd(&counts[s], 1.0f);
  const v4f* src = (const v4f*)(x + (size_t)p * CIN + c0);
  float* dst = sp + (size_t)s * CIN + c0;
#pragma unroll
  for (int v = 0; v < 4; ++v) {
    v4f f = src[v];
#pragma unroll
    for (int q = 0; q < 4; ++q) atomicAdd(&dst[v * 4 + q], f[q]);
  }
}

__global__ void k_pool_norm(float* __restrict__ sp, const float* __restrict__ counts) {
  int i = blockIdx.x * blockDim.x + threadIdx.x;      // NSP*CIN threads
  int s = i >> 7;
  float c = counts[s];
  c = c > 1.0f ? c : 1.0f;
  sp[i] = sp[i] / c;
}

// pack [K x 128] f32 row-major weight into per-lane WMMA-B bf16 layout
// out[((kt*8+nt)*32 + lane)*16 + i] = W[(kt*32 + ((lane>>4)<<4) + i) * 128 + nt*16 + (lane&15)]
__global__ void k_pack_lin(const float* __restrict__ W, __bf16* __restrict__ out, int ktiles) {
  int idx = blockIdx.x * blockDim.x + threadIdx.x;
  int total = ktiles * 8 * 32 * 16;
  if (idx >= total) return;
  int i    = idx & 15;
  int lane = (idx >> 4) & 31;
  int t    = idx >> 9;
  int nt   = t & 7;
  int kt   = t >> 3;
  int n = nt * 16 + (lane & 15);
  int k = kt * 32 + ((lane >> 4) << 4) + i;
  out[idx] = (__bf16)W[k * 128 + n];
}

// pack conv weight Wc[OC][IC][3][3] into B layout with k = ((ky*3+kx)*128 + ic), n = oc
__global__ void k_pack_conv(const float* __restrict__ Wc, __bf16* __restrict__ out) {
  int idx = blockIdx.x * blockDim.x + threadIdx.x;
  const int total = 36 * 8 * 32 * 16;
  if (idx >= total) return;
  int i    = idx & 15;
  int lane = (idx >> 4) & 31;
  int t    = idx >> 9;
  int nt   = t & 7;
  int kt   = t >> 3;
  int oc = nt * 16 + (lane & 15);
  int k  = kt * 32 + ((lane >> 4) << 4) + i;   // 0..1151
  int kyx = k >> 7;                            // 0..8
  int ic  = k & 127;
  int ky = kyx / 3, kx = kyx - ky * 3;
  out[idx] = (__bf16)Wc[((oc * 128 + ic) * 3 + ky) * 3 + kx];
}

__global__ void k_sum3(const float* __restrict__ a, const float* __restrict__ b,
                       const float* __restrict__ c, float* __restrict__ o, int n) {
  int i = blockIdx.x * blockDim.x + threadIdx.x;
  int stride = gridDim.x * blockDim.x;
  for (; i < n; i += stride) o[i] = a[i] + b[i] + c[i];
}

// ---------------- EdgeConv: WMMA GEMM [16 edges x 256] x [256 x 128] + relu + atomic-max ---------

#define ASTRIDE 264   // 16 rows x (256 + 8 pad) bf16; keeps 16B alignment, spreads LDS banks

__global__ __launch_bounds__(128) void k_edgeconv(
    const float* __restrict__ nodes, const int* __restrict__ esrc, const int* __restrict__ edst,
    const __bf16* __restrict__ pw, const float* __restrict__ bias, float* __restrict__ Hout) {
  __shared__ __bf16 A[16 * ASTRIDE];
  __shared__ int ssrc[16], sdst[16];
  const int tid = threadIdx.x;
  const int et  = blockIdx.x * 16;
  if (tid < 16) { ssrc[tid] = esrc[et + tid]; sdst[tid] = edst[et + tid]; }
  __syncthreads();
  {
    // 8 threads per edge, 16 channels each: build cat[x_i, x_j - x_i] in bf16
    int e  = tid >> 3;
    int c0 = (tid & 7) << 4;
    const v4f* xi = (const v4f*)(nodes + (size_t)sdst[e] * CIN + c0);
    const v4f* xj = (const v4f*)(nodes + (size_t)ssrc[e] * CIN + c0);
    __bf16* ai = &A[e * ASTRIDE + c0];
    __bf16* aj = &A[e * ASTRIDE + 128 + c0];
#pragma unroll
    for (int v = 0; v < 4; ++v) {
      v4f fa = xi[v], fb = xj[v];
#pragma unroll
      for (int q = 0; q < 4; ++q) {
        ai[v * 4 + q] = (__bf16)fa[q];
        aj[v * 4 + q] = (__bf16)(fb[q] - fa[q]);
      }
    }
  }
  __syncthreads();

  const int lane  = tid & 31;
  const int wave  = tid >> 5;
  const int nt0   = wave * 2;
  const int row   = lane & 15;
  const int khalf = (lane >> 4) << 4;   // 0 or 16 (contiguous K per lane half)
  v8f acc0 = {}, acc1 = {};
#pragma unroll
  for (int kt = 0; kt < 8; ++kt) {
    int kb = kt * 32 + khalf;
    v8bf c1 = *(const v8bf*)&A[row * ASTRIDE + kb];        // K = kb .. kb+7
    v8bf c2 = *(const v8bf*)&A[row * ASTRIDE + kb + 8];    // K = kb+8 .. kb+15
    v16bf a;
#pragma unroll
    for (int q = 0; q < 8; ++q) { a[q] = c1[q]; a[q + 8] = c2[q]; }
    v16bf b0 = *(const v16bf*)(pw + ((size_t)(kt * 8 + nt0) * 32 + lane) * 16);
    v16bf b1 = *(const v16bf*)(pw + ((size_t)(kt * 8 + nt0 + 1) * 32 + lane) * 16);
    acc0 = __builtin_amdgcn_wmma_f32_16x16x32_bf16(false, a, false, b0, (short)0, acc0, false, false);
    acc1 = __builtin_amdgcn_wmma_f32_16x16x32_bf16(false, a, false, b1, (short)0, acc1, false, false);
  }
  // D layout: VGPR r, lanes 0-15 -> M=r, lanes 16-31 -> M=8+r; N = nt*16 + (lane&15)
  int n0 = nt0 * 16 + (lane & 15);
  int n1 = n0 + 16;
  float bb0 = bias[n0], bb1 = bias[n1];
#pragma unroll
  for (int r = 0; r < 8; ++r) {
    int m = (lane < 16) ? r : (8 + r);
    int d = sdst[m];
    float v0 = acc0[r] + bb0; v0 = v0 > 0.0f ? v0 : 0.0f;
    float v1 = acc1[r] + bb1; v1 = v1 > 0.0f ? v1 : 0.0f;
    // relu output >= 0 -> unsigned-bit compare == float compare; table pre-zeroed
    atomicMax((unsigned int*)&Hout[(size_t)d * OC + n0], __float_as_uint(v0));
    atomicMax((unsigned int*)&Hout[(size_t)d * OC + n1], __float_as_uint(v1));
  }
}

// -------- fused unpool -> 3x3 conv (implicit GEMM, K=1152) -> x Wl + bl -> out ------------------

#define PSTRIDE 136   // 128 + 8 pad bf16 per patch location

__global__ __launch_bounds__(128) void k_conv_linear(
    const float* __restrict__ H4, const int* __restrict__ seg,
    const __bf16* __restrict__ pwc, const float* __restrict__ bc,
    const float* __restrict__ Wl, const float* __restrict__ bl, float* __restrict__ out) {
  __shared__ __bf16 patch[54 * PSTRIDE];   // 3 rows x 18 cols x 128 ch
  __shared__ float convres[16 * 128];
  __shared__ float wl[128 * 16];
  const int tid = threadIdx.x;
  const int y   = blockIdx.x >> 5;
  const int x0  = (blockIdx.x & 31) << 4;

  for (int i = tid; i < 128 * 16; i += 128) wl[i] = Wl[i];

  for (int l = tid; l < 54 * 8; l += 128) {
    int loc = l >> 3;
    int c0  = (l & 7) << 4;
    int pr = loc / 18, pc = loc - pr * 18;
    int yy = y + pr - 1, xx = x0 + pc - 1;
    __bf16* dp = &patch[loc * PSTRIDE + c0];
    if (yy >= 0 && yy < HH && xx >= 0 && xx < WW) {
      int s = seg[yy * WW + xx];
      const v4f* sp4 = (const v4f*)(H4 + (size_t)s * OC + c0);
#pragma unroll
      for (int v = 0; v < 4; ++v) {
        v4f f = sp4[v];
#pragma unroll
        for (int q = 0; q < 4; ++q) dp[v * 4 + q] = (__bf16)f[q];
      }
    } else {
#pragma unroll
      for (int q = 0; q < 16; ++q) dp[q] = (__bf16)0.0f;
    }
  }
  __syncthreads();

  const int lane  = tid & 31;
  const int wave  = tid >> 5;
  const int nt0   = wave * 2;
  const int m     = lane & 15;
  const int khalf = (lane >> 4) << 4;
  v8f acc0 = {}, acc1 = {};
  for (int kt = 0; kt < 36; ++kt) {
    int kyx = kt >> 2;                 // (ky,kx) changes every 4 K-tiles
    int ky = kyx / 3, kx = kyx - ky * 3;
    int icb = (kt & 3) << 5;
    const __bf16* base = &patch[(ky * 18 + m + kx) * PSTRIDE + icb + khalf];
    v8bf c1 = *(const v8bf*)base;
    v8bf c2 = *(const v8bf*)(base + 8);
    v16bf a;
#pragma unroll
    for (int q = 0; q < 8; ++q) { a[q] = c1[q]; a[q + 8] = c2[q]; }
    v16bf b0 = *(const v16bf*)(pwc + ((size_t)(kt * 8 + nt0) * 32 + lane) * 16);
    v16bf b1 = *(const v16bf*)(pwc + ((size_t)(kt * 8 + nt0 + 1) * 32 + lane) * 16);
    acc0 = __builtin_amdgcn_wmma_f32_16x16x32_bf16(false, a, false, b0, (short)0, acc0, false, false);
    acc1 = __builtin_amdgcn_wmma_f32_16x16x32_bf16(false, a, false, b1, (short)0, acc1, false, false);
  }
  int n0 = nt0 * 16 + (lane & 15);
  int n1 = n0 + 16;
  float bb0 = bc[n0], bb1 = bc[n1];
#pragma unroll
  for (int r = 0; r < 8; ++r) {
    int mm = (lane < 16) ? r : (8 + r);
    convres[mm * 128 + n0] = acc0[r] + bb0;
    convres[mm * 128 + n1] = acc1[r] + bb1;
  }
  __syncthreads();

  // tiny final linear 128 -> 16, fused
  for (int w = tid; w < 16 * 16; w += 128) {
    int mm = w >> 4, nc = w & 15;
    float s = bl[nc];
#pragma unroll 8
    for (int oc = 0; oc < 128; ++oc) s += convres[mm * 128 + oc] * wl[oc * 16 + nc];
    out[((size_t)(y * WW + x0 + mm)) * NCLS + nc] = s;
  }
}

// ---------------- host-side orchestration ----------------

extern "C" void kernel_launch(void* const* d_in, const int* in_sizes, int n_in,
                              void* d_out, int out_size, void* d_ws, size_t ws_size,
                              hipStream_t stream) {
  (void)in_sizes; (void)n_in; (void)out_size; (void)ws_size;
  const float* x    = (const float*)d_in[0];
  const int*   eidx = (const int*)d_in[1];
  const int*   seg  = (const int*)d_in[2];
  // d_in[3] = n_sp (constant 16384 per harness)
  const float* W1 = (const float*)d_in[4];
  const float* b1 = (const float*)d_in[5];
  const float* W2 = (const float*)d_in[6];
  const float* b2 = (const float*)d_in[7];
  const float* Wc = (const float*)d_in[8];
  const float* bc = (const float*)d_in[9];
  const float* Wl = (const float*)d_in[10];
  const float* bl = (const float*)d_in[11];
  float* out = (float*)d_out;

  char* ws = (char*)d_ws;
  const size_t HB = (size_t)NSP * OC * sizeof(float);       // 8 MB per node table
  float*  counts = (float*)ws;                               // 16384 f32
  float*  sp     = (float*)(ws + (1 << 16));
  float*  Hf1    = (float*)(ws + (1 << 16) + 1 * HB);
  float*  Hf2    = (float*)(ws + (1 << 16) + 2 * HB);
  float*  Hf3    = (float*)(ws + (1 << 16) + 3 * HB);
  float*  Hf4    = (float*)(ws + (1 << 16) + 4 * HB);
  __bf16* pW1    = (__bf16*)(ws + (1 << 16) + 5 * HB);
  __bf16* pW2    = pW1 + 256 * 128;
  __bf16* pWc    = pW2 + 256 * 128;

  const int* esrc = eidx;        // edge_index[0]
  const int* edst = eidx + NE;   // edge_index[1]

  // mean pool pixels -> superpixels
  k_zero<<<2048, 256, 0, stream>>>(counts, NSP + NSP * CIN);   // counts+sp contiguous
  k_pool_accum<<<(NE * 8) / 256, 256, 0, stream>>>(x, seg, counts, sp);
  k_pool_norm<<<(NSP * CIN) / 256, 256, 0, stream>>>(sp, counts);

  // pack weights into WMMA-B bf16 layouts
  k_pack_lin<<<(8 * 8 * 512) / 256, 256, 0, stream>>>(W1, pW1, 8);
  k_pack_lin<<<(8 * 8 * 512) / 256, 256, 0, stream>>>(W2, pW2, 8);
  k_pack_conv<<<(36 * 8 * 512) / 256, 256, 0, stream>>>(Wc, pWc);

  // three EdgeConv layers (layers 2 and 3 reuse W2/b2, per reference)
  k_zero<<<2048, 256, 0, stream>>>(Hf1, NSP * OC);
  k_edgeconv<<<NE / 16, 128, 0, stream>>>(sp, esrc, edst, pW1, b1, Hf1);
  k_zero<<<2048, 256, 0, stream>>>(Hf2, NSP * OC);
  k_edgeconv<<<NE / 16, 128, 0, stream>>>(Hf1, esrc, edst, pW2, b2, Hf2);
  k_zero<<<2048, 256, 0, stream>>>(Hf3, NSP * OC);
  k_edgeconv<<<NE / 16, 128, 0, stream>>>(Hf2, esrc, edst, pW2, b2, Hf3);
  k_sum3<<<2048, 256, 0, stream>>>(Hf1, Hf2, Hf3, Hf4, NSP * OC);

  // fused unpool + 3x3 conv + final linear
  k_conv_linear<<<HH * (WW / 16), 128, 0, stream>>>(Hf4, seg, pWc, bc, Wl, bl, out);
}
